// AttnModule_27152783245631
// MI455X (gfx1250) — matrix operand order repaired and checked by reading, not compiled, not run on hardware
//
#include <hip/hip_runtime.h>
#include <math.h>

// ---------------------------------------------------------------------------
// CDNA5 (gfx1250) wave32 WMMA implementation of the attention/AdaIN/conv block
// ---------------------------------------------------------------------------

typedef _Float16 h16;
typedef __attribute__((ext_vector_type(16))) _Float16 v16h;
typedef __attribute__((ext_vector_type(8)))  float    v8f;
typedef __attribute__((ext_vector_type(4)))  unsigned int u32x4;
typedef __attribute__((ext_vector_type(8)))  int          i32x8;
typedef __attribute__((ext_vector_type(4)))  int          i32x4;

#define NB   16          // batch
#define CCH  512         // channels
#define CQN  64          // query channels
#define HW   1024        // H*W tokens
#define KTOT (2*CCH*9)   // implicit-GEMM K for 3x3 conv over 2C inputs = 9216

#if defined(__has_builtin)
#if __has_builtin(__builtin_amdgcn_tensor_load_to_lds)
#define HAVE_TDM 1
#endif
#if __has_builtin(__builtin_amdgcn_s_wait_tensorcnt)
#define HAVE_TENSORCNT 1
#endif
#endif

static __device__ __forceinline__ int lane_id() { return (int)(threadIdx.x & 31u); }

static __device__ __forceinline__ v8f wmma_f16(v16h a, v16h b, v8f c) {
  return __builtin_amdgcn_wmma_f32_16x16x32_f16(false, a, false, b, (short)0, c,
                                                false, false);
}

// --- fragment loaders implementing the ISA 7.12.2 wave32 layouts -----------
// A (16x32 f16, MxK): lane&15 = M row; lanes0-15: V0..3=K0..7, V4..7=K16..23
//                     lanes16-31: V0..3=K8..15, V4..7=K24..31
static __device__ __forceinline__ v16h load_a_rm(const h16* A, int ld, int m0, int k0) {
  const int lane = lane_id();
  const long m = (long)(m0 + (lane & 15));
  const int kb = (lane >> 4) ? 8 : 0;
  v16h r;
#pragma unroll
  for (int e = 0; e < 16; ++e) {
    const int j = e >> 1, hh = e & 1;
    const int kk = (j < 4) ? (2 * j + hh + kb) : (16 + 2 * (j - 4) + hh + kb);
    r[e] = A[m * ld + (k0 + kk)];
  }
  return r;
}
// A[m,k] gathered from a column-major source: A[m,k] = P[k*ld + m]
static __device__ __forceinline__ v16h load_a_cm(const h16* P, int ld, int m0, int k0) {
  const int lane = lane_id();
  const int m = m0 + (lane & 15);
  const int kb = (lane >> 4) ? 8 : 0;
  v16h r;
#pragma unroll
  for (int e = 0; e < 16; ++e) {
    const int j = e >> 1, hh = e & 1;
    const int kk = (j < 4) ? (2 * j + hh + kb) : (16 + 2 * (j - 4) + hh + kb);
    r[e] = P[(long)(k0 + kk) * ld + m];
  }
  return r;
}
// B (32x16 f16, KxN): lanes0-15: V0..3=K0..7, V4..7=K16..23 ; lanes16-31: V0..3=K8..15, V4..7=K24..31
static __device__ __forceinline__ v16h load_b_rm(const h16* B, int ld, int k0, int n0) {
  const int lane = lane_id();
  const int n = n0 + (lane & 15);
  const int kb = (lane >> 4) ? 8 : 0;
  v16h r;
#pragma unroll
  for (int e = 0; e < 16; ++e) {
    const int j = e >> 1, hh = e & 1;
    const int kk = (j < 4) ? (2 * j + hh + kb) : (16 + 2 * (j - 4) + hh + kb);
    r[e] = B[(long)(k0 + kk) * ld + n];
  }
  return r;
}
// B[k,n] gathered from row-major [n][k] source (transposed operand)
static __device__ __forceinline__ v16h load_b_cm(const h16* P, int ld, int k0, int n0) {
  const int lane = lane_id();
  const long n = (long)(n0 + (lane & 15));
  const int kb = (lane >> 4) ? 8 : 0;
  v16h r;
#pragma unroll
  for (int e = 0; e < 16; ++e) {
    const int j = e >> 1, hh = e & 1;
    const int kk = (j < 4) ? (2 * j + hh + kb) : (16 + 2 * (j - 4) + hh + kb);
    r[e] = P[n * ld + (k0 + kk)];
  }
  return r;
}

// ---------------------------------------------------------------------------
// 1) fused stats + AdaIN + f32->f16 activation conversion (one block per b,c)
// ---------------------------------------------------------------------------
__global__ __launch_bounds__(256)
void k_prep(const float* __restrict__ inp, const float* __restrict__ x,
            const float* __restrict__ y, h16* __restrict__ x_h,
            h16* __restrict__ y_h, h16* __restrict__ ad_h,
            float* __restrict__ xmean, float* __restrict__ xstd,
            float* __restrict__ imean, float* __restrict__ istd,
            float* __restrict__ rowsum) {
  const int bc = blockIdx.x;                      // b*512 + c
  const float* xr = x + (long)bc * HW;
  const float* ir = inp + (long)bc * HW;
  const float* yr = y + (long)bc * HW;
  __shared__ float red[256];
  auto blockReduce = [&](float v) -> float {
    red[threadIdx.x] = v; __syncthreads();
    for (int s = 128; s > 0; s >>= 1) {
      if ((int)threadIdx.x < s) red[threadIdx.x] += red[threadIdx.x + s];
      __syncthreads();
    }
    float r = red[0]; __syncthreads();
    return r;
  };
  float sx = 0.f, sxx = 0.f, si = 0.f, sii = 0.f;
  for (int i = threadIdx.x; i < HW; i += 256) {
    const float xv = xr[i]; sx += xv; sxx += xv * xv;
    const float iv = ir[i]; si += iv; sii += iv * iv;
    x_h[(long)bc * HW + i] = (h16)xv;
    y_h[(long)bc * HW + i] = (h16)yr[i];
  }
  sx = blockReduce(sx); sxx = blockReduce(sxx);
  si = blockReduce(si); sii = blockReduce(sii);
  const float mx  = sx / HW;
  const float sdx = sqrtf((sxx - HW * mx * mx) / (HW - 1) + 1e-5f);
  const float mi  = si / HW;
  const float sdi = sqrtf((sii - HW * mi * mi) / (HW - 1) + 1e-5f);
  if (threadIdx.x == 0) { xmean[bc] = mx; xstd[bc] = sdx; imean[bc] = mi; istd[bc] = sdi; }
  const float inv = 1.0f / sdi;
  float ssum = 0.f;
  for (int i = threadIdx.x; i < HW; i += 256) {
    const float a = (ir[i] - mi) * inv * sdx + mx;
    ad_h[(long)bc * HW + i] = (h16)a;
    ssum += a;
  }
  ssum = blockReduce(ssum);
  if (threadIdx.x == 0) rowsum[bc] = ssum;
}

// ---------------------------------------------------------------------------
// 2) f32 -> f16 weight conversion
// ---------------------------------------------------------------------------
__global__ void k_cvt(const float* __restrict__ s, h16* __restrict__ d, int n) {
  const int i = blockIdx.x * 256 + threadIdx.x;
  if (i < n) d[i] = (h16)s[i];
}

// ---------------------------------------------------------------------------
// 3) batched 1x1-conv projection GEMM: out[b,o,n] = sum_c W[o,c]*X[b,c,n]+bias
// ---------------------------------------------------------------------------
__global__ __launch_bounds__(256)
void k_proj(const h16* __restrict__ W, const float* __restrict__ bias,
            const h16* __restrict__ X, h16* __restrict__ out, int M, int K) {
  const int w = threadIdx.x >> 5;
  const int n0 = (blockIdx.x * 8 + w) * 16;
  const int m0 = blockIdx.y * 16;
  const int b  = blockIdx.z;
  const h16* Xb = X + (long)b * K * HW;
  v8f acc = {};
  for (int k0 = 0; k0 < K; k0 += 32) {
    const v16h a  = load_a_rm(W, K, m0, k0);
    const v16h bf = load_b_rm(Xb, HW, k0, n0);
    acc = wmma_f16(a, bf, acc);
  }
  h16* ob = out + (long)b * M * HW;
  const int lane = lane_id();
  const int nn = n0 + (lane & 15);
  const int mb = (lane >> 4) ? 8 : 0;
#pragma unroll
  for (int r = 0; r < 8; ++r) {
    const int mm = m0 + mb + r;
    ob[(long)mm * HW + nn] = (h16)(acc[r] + bias[mm]);
  }
}

// ---------------------------------------------------------------------------
// 4) attention: S = (pq^T pk)/t ; softmax rows ; f16 attn out
// ---------------------------------------------------------------------------
__global__ __launch_bounds__(32)
void k_attn(const h16* __restrict__ PQ, const h16* __restrict__ PK,
            h16* __restrict__ ATT) {
  const int b  = blockIdx.y;
  const int m0 = blockIdx.x * 16;
  const h16* qp = PQ + (long)b * CQN * HW;
  const h16* kp = PK + (long)b * CQN * HW;
  h16* att = ATT + (long)b * HW * HW;
  const float inv_t = 0.125f;                    // 1/sqrt(64)
  const v16h a0 = load_a_cm(qp, HW, m0, 0);      // K=64 hoisted A fragments
  const v16h a1 = load_a_cm(qp, HW, m0, 32);
  const int lane = lane_id();
  const int nn_lo = lane & 15;
  const int hi = lane >> 4;
  float rmax[8];
#pragma unroll
  for (int r = 0; r < 8; ++r) rmax[r] = -3.0e38f;
  for (int kt = 0; kt < 64; ++kt) {              // pass 1: row max
    const int n0 = kt * 16;
    v8f acc = {};
    acc = wmma_f16(a0, load_b_rm(kp, HW, 0, n0), acc);
    acc = wmma_f16(a1, load_b_rm(kp, HW, 32, n0), acc);
#pragma unroll
    for (int r = 0; r < 8; ++r) rmax[r] = fmaxf(rmax[r], acc[r] * inv_t);
  }
#pragma unroll
  for (int r = 0; r < 8; ++r)
#pragma unroll
    for (int off = 8; off > 0; off >>= 1)
      rmax[r] = fmaxf(rmax[r], __shfl_xor(rmax[r], off, 32));
  float rsum[8];
#pragma unroll
  for (int r = 0; r < 8; ++r) rsum[r] = 0.f;
  for (int kt = 0; kt < 64; ++kt) {              // pass 2: exp, store, sum
    const int n0 = kt * 16;
    v8f acc = {};
    acc = wmma_f16(a0, load_b_rm(kp, HW, 0, n0), acc);
    acc = wmma_f16(a1, load_b_rm(kp, HW, 32, n0), acc);
#pragma unroll
    for (int r = 0; r < 8; ++r) {
      const float e = __expf(acc[r] * inv_t - rmax[r]);
      rsum[r] += e;
      const int mm = m0 + r + (hi ? 8 : 0);
      att[(long)mm * HW + (n0 + nn_lo)] = (h16)e;
    }
  }
#pragma unroll
  for (int r = 0; r < 8; ++r)
#pragma unroll
    for (int off = 8; off > 0; off >>= 1)
      rsum[r] += __shfl_xor(rsum[r], off, 32);
  float invs[16];
#pragma unroll
  for (int r = 0; r < 8; ++r) {
    invs[r]     = 1.0f / __shfl(rsum[r], 0, 32);
    invs[8 + r] = 1.0f / __shfl(rsum[r], 16, 32);
  }
  asm volatile("s_wait_storecnt 0x0" ::: "memory");  // our stores -> our loads
  h16* base = att + (long)m0 * HW;
  for (int i = lane; i < 16 * HW / 2; i += 32) {     // pass 3: normalize
    const int row = i >> 9;
    const float v0 = (float)base[2 * i];
    const float v1 = (float)base[2 * i + 1];
    base[2 * i]     = (h16)(v0 * invs[row]);
    base[2 * i + 1] = (h16)(v1 * invs[row]);
  }
}

// ---------------------------------------------------------------------------
// 5) value GEMMs, 16x64 per wave (A fragment reused across 4 WMMA), fused
//    epilogues into cat[b, 2C, N]
// ---------------------------------------------------------------------------
__global__ __launch_bounds__(256)
void k_vgemm(const h16* __restrict__ A, const h16* __restrict__ ATT,
             h16* __restrict__ cat, const float* __restrict__ xf,
             const float* __restrict__ inpf, const float* __restrict__ xmean,
             const float* __restrict__ xstd, const float* __restrict__ imean,
             const float* __restrict__ istd, const float* __restrict__ rowsum,
             const float* __restrict__ alpha, const float* __restrict__ beta,
             int mode) {
  const int w = threadIdx.x >> 5;
  const int tile = blockIdx.x * 8 + w;   // 512 tiles = 32 m-tiles x 16 n-groups
  const int m0 = (tile >> 4) * 16;
  const int ng = (tile & 15) * 64;
  const int b = blockIdx.z;
  const h16* Ab  = A + (long)b * CCH * HW;
  const h16* att = ATT + (long)b * HW * HW;
  v8f acc[4] = {};
  for (int k0 = 0; k0 < HW; k0 += 32) {
    const v16h a = load_a_rm(Ab, HW, m0, k0);
#pragma unroll
    for (int t = 0; t < 4; ++t) {
      const v16h bf = load_b_cm(att, HW, k0, ng + 16 * t);
      acc[t] = wmma_f16(a, bf, acc[t]);
    }
  }
  const int lane = lane_id();
  const int mb = (lane >> 4) ? 8 : 0;
  if (mode == 0) {
    const float bet = beta[0];
    h16* dst = cat + (long)b * (2 * CCH) * HW;
#pragma unroll
    for (int t = 0; t < 4; ++t) {
      const int nn = ng + 16 * t + (lane & 15);
#pragma unroll
      for (int r = 0; r < 8; ++r) {
        const int c = m0 + mb + r;
        const float xv = xf[((long)b * CCH + c) * HW + nn];
        dst[(long)c * HW + nn] = (h16)((1.f - bet) * xv + bet * acc[t][r]);
      }
    }
  } else {
    const float al = alpha[0];
    h16* dst = cat + (long)b * (2 * CCH) * HW + (long)CCH * HW;
#pragma unroll
    for (int t = 0; t < 4; ++t) {
      const int nn = ng + 16 * t + (lane & 15);
#pragma unroll
      for (int r = 0; r < 8; ++r) {
        const int c = m0 + mb + r;
        const int bc = b * CCH + c;
        const float rev = rowsum[bc] - acc[t][r];
        const float ad =
            (inpf[(long)bc * HW + nn] - imean[bc]) / istd[bc] * xstd[bc] + xmean[bc];
        dst[(long)c * HW + nn] = (h16)((1.f - al) * rev + al * ad);
      }
    }
  }
}

// ---------------------------------------------------------------------------
// 6) 3x3 conv (2C->C, SAME) as implicit GEMM. Block tile 128x64, wave tile
//    32x32 (4 WMMA / K-step). Weight panel staged by the Tensor Data Mover
//    (tensor_load_to_lds + s_wait_tensorcnt) with TDM LDS padding matching
//    the [128][34] layout; im2col panel gathered manually. Bias + LeakyReLU.
// ---------------------------------------------------------------------------
__global__ __launch_bounds__(256)
void k_conv(const h16* __restrict__ Wc_h, const h16* __restrict__ cat,
            const float* __restrict__ bc_, float* __restrict__ out) {
  __shared__ h16 As[128][34];   // 32 halves (=16 DW) + 1 DW pad per row
  __shared__ h16 Bs[32][66];
  const int b  = blockIdx.z;
  const int m0 = blockIdx.y * 128;
  const int n0 = blockIdx.x * 64;
  const int tid = threadIdx.x;
  const int w = tid >> 5, lane = tid & 31;
  const int mw = (w & 3) * 32, nw = (w >> 2) * 32;
  const h16* catb = cat + (long)b * (2 * CCH) * HW;
  v8f acc[2][2] = {};
#if defined(HAVE_TDM)
  // D# group1 (invariant): data_size=2B, pad_enable, pad_interval=16DW,
  // pad_amount=1DW, tensor_dim0=K=9216, tensor_dim1=512 rows, tile 32x128,
  // tensor_dim0_stride=9216.
  i32x8 g1;
  g1[0] = (int)((1u << 16) | (1u << 20) | (3u << 22));
  g1[1] = (int)((unsigned)KTOT << 16);
  g1[2] = (int)(512u << 16);
  g1[3] = (int)(32u << 16);
  g1[4] = 128;
  g1[5] = KTOT;
  g1[6] = 0;
  g1[7] = 0;
  const i32x4 gz = {0, 0, 0, 0};
  const unsigned ldsAs = (unsigned)(unsigned long long)(void*)&As[0][0];
#endif
  for (int k0 = 0; k0 < KTOT; k0 += 32) {
    if (k0 + 32 < KTOT)
      __builtin_prefetch(&Wc_h[(long)(m0 + (tid >> 2)) * KTOT + k0 + 32], 0, 0);
#if defined(HAVE_TDM)
    if (w == 0) {   // one wave drives the TDM for the whole 128x32 panel
      const unsigned long long ga =
          (unsigned long long)(const void*)&Wc_h[(long)m0 * KTOT + k0];
      u32x4 g0;
      g0[0] = 1u;                                   // count=1 valid descriptor
      g0[1] = ldsAs;                                // LDS byte address
      g0[2] = (unsigned)ga;                         // global addr lo
      g0[3] = (unsigned)((ga >> 32) & 0x01FFFFFFu) | (2u << 30);  // hi | type=2
#if __clang_major__ >= 23
      const i32x8 gz8 = {0, 0, 0, 0, 0, 0, 0, 0};
      __builtin_amdgcn_tensor_load_to_lds(g0, g1, gz, gz, gz8, 0);
#else
      __builtin_amdgcn_tensor_load_to_lds(g0, g1, gz, gz, 0);
#endif
    }
#else
    for (int idx = tid; idx < 128 * 32; idx += 256) {   // fallback staging
      const int r = idx >> 5, kk = idx & 31;
      As[r][kk] = Wc_h[(long)(m0 + r) * KTOT + k0 + kk];
    }
#endif
    for (int idx = tid; idx < 32 * 64; idx += 256) {    // gather im2col panel
      const int kk = idx >> 6, nn = idx & 63;
      const int k = k0 + kk;
      const int ic = k / 9, t = k - ic * 9;
      const int dy = t / 3 - 1, dx = t % 3 - 1;
      const int n = n0 + nn;
      const int yy = (n >> 5) + dy, xx = (n & 31) + dx;
      h16 v = (h16)0.0f;
      if (yy >= 0 && yy < 32 && xx >= 0 && xx < 32)
        v = catb[(long)ic * HW + yy * 32 + xx];
      Bs[kk][nn] = v;
    }
#if defined(HAVE_TDM)
    if (w == 0) {
#if defined(HAVE_TENSORCNT)
      __builtin_amdgcn_s_wait_tensorcnt(0);
#else
      asm volatile("s_wait_tensorcnt 0x0" ::: "memory");
#endif
    }
#endif
    __syncthreads();
    const v16h a0 = load_a_rm(&As[0][0], 34, mw, 0);
    const v16h a1 = load_a_rm(&As[0][0], 34, mw + 16, 0);
    const v16h b0 = load_b_rm(&Bs[0][0], 66, 0, nw);
    const v16h b1 = load_b_rm(&Bs[0][0], 66, 0, nw + 16);
    acc[0][0] = wmma_f16(a0, b0, acc[0][0]);
    acc[0][1] = wmma_f16(a0, b1, acc[0][1]);
    acc[1][0] = wmma_f16(a1, b0, acc[1][0]);
    acc[1][1] = wmma_f16(a1, b1, acc[1][1]);
    __syncthreads();
  }
  const int mb = (lane >> 4) ? 8 : 0;
#pragma unroll
  for (int tm = 0; tm < 2; ++tm)
#pragma unroll
    for (int tn = 0; tn < 2; ++tn) {
      const int nn = n0 + nw + tn * 16 + (lane & 15);
#pragma unroll
      for (int r = 0; r < 8; ++r) {
        const int o = m0 + mw + tm * 16 + mb + r;
        float v = acc[tm][tn][r] + bc_[o];
        v = (v >= 0.f) ? v : 0.2f * v;
        out[((long)b * CCH + o) * HW + nn] = v;
      }
    }
}

// ---------------------------------------------------------------------------
// host launcher
// ---------------------------------------------------------------------------
extern "C" void kernel_launch(void* const* d_in, const int* in_sizes, int n_in,
                              void* d_out, int out_size, void* d_ws, size_t ws_size,
                              hipStream_t stream) {
  const float* inp = (const float*)d_in[0];
  const float* x   = (const float*)d_in[1];
  const float* y   = (const float*)d_in[2];
  const float* Wq  = (const float*)d_in[3];
  const float* bq  = (const float*)d_in[4];
  const float* Wk  = (const float*)d_in[5];
  const float* bk  = (const float*)d_in[6];
  const float* Wv  = (const float*)d_in[7];
  const float* bv  = (const float*)d_in[8];
  const float* Wc  = (const float*)d_in[9];
  const float* bc  = (const float*)d_in[10];
  const float* alpha = (const float*)d_in[11];
  const float* beta  = (const float*)d_in[12];
  float* out = (float*)d_out;

  size_t off = 0;
  auto carve = [&](size_t bytes) -> void* {
    void* p = (char*)d_ws + off;
    off += (bytes + 255) & ~(size_t)255;
    return p;
  };
  const size_t actB = (size_t)NB * CCH * HW * sizeof(h16);   // 16 MB
  const size_t prjB = (size_t)NB * CQN * HW * sizeof(h16);   // 2 MB
  const size_t attB = (size_t)NB * HW * HW * sizeof(h16);    // 32 MB
  h16* x_h   = (h16*)carve(actB);
  h16* y_h   = (h16*)carve(actB);
  h16* ad_h  = (h16*)carve(actB);
  h16* pv_h  = (h16*)carve(actB);
  h16* pq_h  = (h16*)carve(prjB);
  h16* pk_h  = (h16*)carve(prjB);
  h16* pqp_h = (h16*)carve(prjB);
  h16* pkp_h = (h16*)carve(prjB);
  h16* attI  = (h16*)carve(attB);
  h16* attP  = (h16*)carve(attB);
  h16* cat_h = (h16*)carve(attB);                            // [B,2C,N] = 32MB
  h16* Wq_h  = (h16*)carve((size_t)CQN * CCH * sizeof(h16));
  h16* Wk_h  = (h16*)carve((size_t)CQN * CCH * sizeof(h16));
  h16* Wv_h  = (h16*)carve((size_t)CCH * CCH * sizeof(h16));
  h16* Wc_h  = (h16*)carve((size_t)CCH * KTOT * sizeof(h16));
  float* xmean = (float*)carve(NB * CCH * sizeof(float));
  float* xstd  = (float*)carve(NB * CCH * sizeof(float));
  float* imean = (float*)carve(NB * CCH * sizeof(float));
  float* istd  = (float*)carve(NB * CCH * sizeof(float));
  float* rowsum = (float*)carve(NB * CCH * sizeof(float));
  (void)ws_size; (void)in_sizes; (void)n_in; (void)out_size;

  // 1) stats + AdaIN + f16 activation conversion
  k_prep<<<dim3(NB * CCH), dim3(256), 0, stream>>>(inp, x, y, x_h, y_h, ad_h,
                                                   xmean, xstd, imean, istd, rowsum);
  // 2) weight conversions
  auto cvt = [&](const float* s, h16* d, int n) {
    k_cvt<<<dim3((n + 255) / 256), dim3(256), 0, stream>>>(s, d, n);
  };
  cvt(Wq, Wq_h, CQN * CCH);
  cvt(Wk, Wk_h, CQN * CCH);
  cvt(Wv, Wv_h, CCH * CCH);
  cvt(Wc, Wc_h, CCH * KTOT);
  // 3) projections
  k_proj<<<dim3(8, CQN / 16, NB), dim3(256), 0, stream>>>(Wq_h, bq, x_h, pq_h, CQN, CCH);
  k_proj<<<dim3(8, CQN / 16, NB), dim3(256), 0, stream>>>(Wk_h, bk, y_h, pk_h, CQN, CCH);
  k_proj<<<dim3(8, CQN / 16, NB), dim3(256), 0, stream>>>(Wq_h, bq, y_h, pqp_h, CQN, CCH);
  k_proj<<<dim3(8, CQN / 16, NB), dim3(256), 0, stream>>>(Wk_h, bk, x_h, pkp_h, CQN, CCH);
  k_proj<<<dim3(8, CCH / 16, NB), dim3(256), 0, stream>>>(Wv_h, bv, x_h, pv_h, CCH, CCH);
  // 4) softmax attentions
  k_attn<<<dim3(HW / 16, NB), dim3(32), 0, stream>>>(pq_h, pk_h, attI);
  k_attn<<<dim3(HW / 16, NB), dim3(32), 0, stream>>>(pqp_h, pkp_h, attP);
  // 5) value GEMMs + fused epilogues into cat
  k_vgemm<<<dim3(64, 1, NB), dim3(256), 0, stream>>>(
      pv_h, attI, cat_h, x, inp, xmean, xstd, imean, istd, rowsum, alpha, beta, 0);
  k_vgemm<<<dim3(64, 1, NB), dim3(256), 0, stream>>>(
      ad_h, attP, cat_h, x, inp, xmean, xstd, imean, istd, rowsum, alpha, beta, 1);
  // 6) 3x3 conv + bias + LeakyReLU
  k_conv<<<dim3(HW / 64, CCH / 128, NB), dim3(256), 0, stream>>>(Wc_h, cat_h, bc, out);
}
